// RingGIN_10247791968545
// MI455X (gfx1250) — matrix operand order repaired and checked by reading, not compiled, not run on hardware
//
#include <hip/hip_runtime.h>
#include <hip/hip_bf16.h>

typedef __attribute__((ext_vector_type(16))) __bf16 v16bf;
typedef __attribute__((ext_vector_type(8)))  __bf16 v8bf;
typedef __attribute__((ext_vector_type(8)))  float  v8f;

#define DDIM 128
#define LDS_PITCH (DDIM + 8)   // pad 16B per row -> conflict-free ds_load_b128

// ---------------------------------------------------------------------------
// Convert fp32 -> bf16 (used once per launch for the 9 weight matrices).
// ---------------------------------------------------------------------------
__global__ __launch_bounds__(256) void gin_f32_to_bf16_kernel(
    const float* __restrict__ in, __bf16* __restrict__ out, int n)
{
    int i = blockIdx.x * blockDim.x + threadIdx.x;
    if (i < n) out[i] = (__bf16)in[i];
}

// ---------------------------------------------------------------------------
// GEMM: Y[N x 128] = A[N x 128] @ W[128 x 128]^T + bias
//   Y[m, n] = sum_d A[m, d] * W[n, d] + bias[n]
// Block = 256 threads = 8 waves; block computes 16 rows x 128 cols.
// A tile staged once into LDS as bf16 (conversion out of the K-loop, 8x less
// global A traffic); W pre-converted to bf16 in workspace (contiguous 32B
// B-fragment loads). v_wmma_f32_16x16x32_bf16, f32 accumulation.
// Full-tile fast-path epilogue: 8 unguarded b32 stores at immediate offsets.
// ---------------------------------------------------------------------------
__global__ __launch_bounds__(256) void gin_gemm_bias_kernel(
    const float* __restrict__ A, const __bf16* __restrict__ Wb,
    const float* __restrict__ bias, float* __restrict__ Y, int N)
{
    __shared__ __bf16 sA[16 * LDS_PITCH];

    const int tid   = threadIdx.x;
    const int tileM = blockIdx.x * 16;

    // ---- Stage A tile (16 x 128 fp32 -> bf16 in LDS), 8 elems per thread ----
    {
        const int row = tid >> 4;           // 0..15
        const int c0  = (tid & 15) * 8;     // 0,8,...,120
        int gr = tileM + row;
        if (gr >= N) gr = N - 1;            // clamp (partial tile safety)
        const float* __restrict__ src = A + (size_t)gr * DDIM + c0;
        const float4 f0 = ((const float4*)src)[0];
        const float4 f1 = ((const float4*)src)[1];
        v8bf pk;
        pk[0] = (__bf16)f0.x; pk[1] = (__bf16)f0.y;
        pk[2] = (__bf16)f0.z; pk[3] = (__bf16)f0.w;
        pk[4] = (__bf16)f1.x; pk[5] = (__bf16)f1.y;
        pk[6] = (__bf16)f1.z; pk[7] = (__bf16)f1.w;
        *(v8bf*)(sA + row * LDS_PITCH + c0) = pk;   // 16B ds_store
    }
    __syncthreads();

    const int wave = tid >> 5;              // 0..7 -> 16-col tile
    const int lane = tid & 31;
    const int hl   = lane >> 4;             // lane half (0/1)
    const int l16  = lane & 15;
    const int tileN = wave * 16;

    const __bf16* __restrict__ arow = sA + l16 * LDS_PITCH;              // A row m (LDS)
    const __bf16* __restrict__ wrow = Wb + (size_t)(tileN + l16) * DDIM; // W row n

    v8f acc = {};
#pragma unroll
    for (int k0 = 0; k0 < DDIM; k0 += 32) {
        // A 16x32 bf16 fragment: elems 0..15 -> k0+16*hl+e (contiguous)
        const v8bf alo = *(const v8bf*)(arow + k0 + 16 * hl);       // ds_load_b128
        const v8bf ahi = *(const v8bf*)(arow + k0 + 16 * hl + 8);   // ds_load_b128
        v16bf afrag;
#pragma unroll
        for (int e = 0; e < 8; ++e) { afrag[e] = alo[e]; afrag[e + 8] = ahi[e]; }
        // B 32x16 bf16 fragment: lane holds col n=l16, k = k0+16*hl+e (contig 32B)
        const v16bf bfrag = *(const v16bf*)(wrow + k0 + 16 * hl);
        acc = __builtin_amdgcn_wmma_f32_16x16x32_bf16(
            /*neg_a=*/false, afrag, /*neg_b=*/false, bfrag,
            /*c_mod=*/(short)0, acc, /*reuse_a=*/false, /*reuse_b=*/false);
    }

    // C/D layout: VGPR r -> row tileM + 8*hl + r, col tileN + l16.
    const float bn = bias[tileN + l16];
    float* __restrict__ ybase =
        Y + (size_t)(tileM + 8 * hl) * DDIM + tileN + l16;
    if (tileM + 16 <= N) {
        // Full tile (always, for N % 16 == 0): branch-free immediate-offset stores.
#pragma unroll
        for (int r = 0; r < 8; ++r) ybase[(size_t)r * DDIM] = acc[r] + bn;
    } else {
#pragma unroll
        for (int r = 0; r < 8; ++r) {
            if (tileM + 8 * hl + r < N) ybase[(size_t)r * DDIM] = acc[r] + bn;
        }
    }
}

// ---------------------------------------------------------------------------
// GIN aggregation: agg[dst] += x[src] (agg pre-initialized to x => h = x + agg)
// One wave per edge, float4 per lane -> 4 dword atomics per lane. L2-resident.
// ---------------------------------------------------------------------------
__global__ __launch_bounds__(256) void gin_scatter_add_kernel(
    const float* __restrict__ x, const int* __restrict__ srcv,
    const int* __restrict__ dstv, float* __restrict__ agg, int E)
{
    const int wavesPerBlock = blockDim.x >> 5;
    const int edge = blockIdx.x * wavesPerBlock + (threadIdx.x >> 5);
    if (edge >= E) return;
    const int lane = threadIdx.x & 31;
    const int s = srcv[edge];
    const int d = dstv[edge];
    const float4 v = ((const float4*)(x + (size_t)s * DDIM))[lane];
    float* ad = agg + (size_t)d * DDIM + lane * 4;
    atomicAdd(ad + 0, v.x);
    atomicAdd(ad + 1, v.y);
    atomicAdd(ad + 2, v.z);
    atomicAdd(ad + 3, v.w);
}

// ---------------------------------------------------------------------------
// BatchNorm statistics: stats[0..127] = sum per column, stats[128..255] = sumsq
// ---------------------------------------------------------------------------
__global__ __launch_bounds__(256) void gin_bn_zero_kernel(float* __restrict__ stats)
{
    stats[threadIdx.x] = 0.0f;
}

__global__ __launch_bounds__(256) void gin_bn_stats_kernel(
    const float* __restrict__ z, int N, int rowsPerBlock,
    float* __restrict__ stats)
{
    const int col  = threadIdx.x & 127;
    const int rsub = threadIdx.x >> 7;      // 0/1
    int r0 = blockIdx.x * rowsPerBlock;
    int r1 = r0 + rowsPerBlock;
    if (r1 > N) r1 = N;
    float s = 0.0f, ss = 0.0f;
    for (int r = r0 + rsub; r < r1; r += 2) {
        float v = z[(size_t)r * DDIM + col];
        s += v;
        ss += v * v;
    }
    __shared__ float sh[256];
    sh[threadIdx.x] = s;
    __syncthreads();
    if (rsub == 0) atomicAdd(&stats[col], sh[col] + sh[col + 128]);
    __syncthreads();
    sh[threadIdx.x] = ss;
    __syncthreads();
    if (rsub == 0) atomicAdd(&stats[128 + col], sh[col] + sh[col + 128]);
}

// ---------------------------------------------------------------------------
// Apply BN (train mode, biased var) + ReLU:  out = relu(g*(z-m)*rsqrt(v+eps)+b)
// ---------------------------------------------------------------------------
__global__ __launch_bounds__(256) void gin_bn_apply_relu_kernel(
    const float* __restrict__ z, const float* __restrict__ stats,
    const float* __restrict__ g, const float* __restrict__ be,
    float* __restrict__ out, int N)
{
    const size_t total = (size_t)N * DDIM;
    size_t i = (size_t)blockIdx.x * blockDim.x + threadIdx.x;
    if (i >= total) return;
    const int col = (int)(i & (DDIM - 1));
    const float invN = 1.0f / (float)N;
    const float mean = stats[col] * invN;
    float var = stats[128 + col] * invN - mean * mean;
    if (var < 0.0f) var = 0.0f;
    const float sc = g[col] * __frsqrt_rn(var + 1e-5f);
    const float sf = be[col] - mean * sc;
    float v = z[i] * sc + sf;
    out[i] = v > 0.0f ? v : 0.0f;
}

// ---------------------------------------------------------------------------
// Output head: out[m, c] = dot(x[mask[m], :], W_out[c, :]) + b_out[c]
// M x C = 100k dots of length 128 — negligible work, C=10 not WMMA-shaped.
// ---------------------------------------------------------------------------
__global__ __launch_bounds__(256) void gin_out_kernel(
    const float* __restrict__ x, const int* __restrict__ mask,
    const float* __restrict__ Wout, const float* __restrict__ bout,
    float* __restrict__ out, int M, int C)
{
    const int idx = blockIdx.x * blockDim.x + threadIdx.x;
    if (idx >= M * C) return;
    const int m = idx / C;
    const int c = idx - m * C;
    const float* __restrict__ xr = x + (size_t)mask[m] * DDIM;
    const float* __restrict__ wr = Wout + (size_t)c * DDIM;
    float acc = bout[c];
#pragma unroll 4
    for (int d = 0; d < DDIM; ++d) acc += xr[d] * wr[d];
    out[idx] = acc;
}

// ---------------------------------------------------------------------------
extern "C" void kernel_launch(void* const* d_in, const int* in_sizes, int n_in,
                              void* d_out, int out_size, void* d_ws, size_t ws_size,
                              hipStream_t stream) {
    const float* x_in   = (const float*)d_in[0];
    const int*   eidx   = (const int*)d_in[1];
    const int*   mask   = (const int*)d_in[2];
    const float* W_init = (const float*)d_in[3];
    const float* b_init = (const float*)d_in[4];
    const float* W1s    = (const float*)d_in[5];
    const float* b1s    = (const float*)d_in[6];
    const float* g1s    = (const float*)d_in[7];
    const float* be1s   = (const float*)d_in[8];
    const float* W2s    = (const float*)d_in[9];
    const float* b2s    = (const float*)d_in[10];
    const float* g2s    = (const float*)d_in[11];
    const float* be2s   = (const float*)d_in[12];
    const float* W_out  = (const float*)d_in[13];
    const float* b_out  = (const float*)d_in[14];

    const int N = in_sizes[0] / DDIM;
    const int E = in_sizes[1] / 2;
    const int M = in_sizes[2];
    const int L = in_sizes[5] / (DDIM * DDIM);
    const int C = in_sizes[13] / DDIM;

    const int* srcv = eidx;       // edge_index[0]
    const int* dstv = eidx + E;   // edge_index[1]

    const int WMAT = DDIM * DDIM;               // 16384 elems per weight matrix
    const size_t mat = (size_t)N * DDIM;
    float*  buf0   = (float*)d_ws;              // current node features x
    float*  buf1   = buf0 + mat;                // agg / h (MLP input)
    float*  buf2   = buf1 + mat;                // pre-BN linear output z
    float*  stats  = buf2 + mat;                // 256 floats: col sums + sumsq
    __bf16* wbInit = (__bf16*)(stats + 256);    // bf16 weights: init, W1s[L], W2s[L]
    __bf16* wb1    = wbInit + WMAT;
    __bf16* wb2    = wb1 + (size_t)L * WMAT;

    const int gemmGrid    = (N + 15) / 16;
    const int scatterGrid = (E + 7) / 8;        // 8 waves/block, 1 edge/wave
    const int rowsPerBlk  = 512;
    const int statsGrid   = (N + rowsPerBlk - 1) / rowsPerBlk;
    const int ewGrid      = (int)((mat + 255) / 256);

    // ---- Pre-convert all weights to bf16 (once per launch, 288 KB total) ----
    gin_f32_to_bf16_kernel<<<(WMAT + 255) / 256, 256, 0, stream>>>(W_init, wbInit, WMAT);
    gin_f32_to_bf16_kernel<<<(L * WMAT + 255) / 256, 256, 0, stream>>>(W1s, wb1, L * WMAT);
    gin_f32_to_bf16_kernel<<<(L * WMAT + 255) / 256, 256, 0, stream>>>(W2s, wb2, L * WMAT);

    // x = x @ W_init^T + b_init
    gin_gemm_bias_kernel<<<gemmGrid, 256, 0, stream>>>(x_in, wbInit, b_init, buf0, N);

    for (int i = 0; i < L; ++i) {
        // h = x + segment_sum(x[src], dst):  buf1 = x, then scatter-add
        hipMemcpyAsync(buf1, buf0, mat * sizeof(float), hipMemcpyDeviceToDevice, stream);
        gin_scatter_add_kernel<<<scatterGrid, 256, 0, stream>>>(buf0, srcv, dstv, buf1, E);

        // z = h @ W1^T + b1 ; h = relu(bn(z))
        gin_gemm_bias_kernel<<<gemmGrid, 256, 0, stream>>>(
            buf1, wb1 + (size_t)i * WMAT, b1s + i * DDIM, buf2, N);
        gin_bn_zero_kernel<<<1, 256, 0, stream>>>(stats);
        gin_bn_stats_kernel<<<statsGrid, 256, 0, stream>>>(buf2, N, rowsPerBlk, stats);
        gin_bn_apply_relu_kernel<<<ewGrid, 256, 0, stream>>>(
            buf2, stats, g1s + i * DDIM, be1s + i * DDIM, buf1, N);

        // z = h @ W2^T + b2 ; x = relu(bn(z))
        gin_gemm_bias_kernel<<<gemmGrid, 256, 0, stream>>>(
            buf1, wb2 + (size_t)i * WMAT, b2s + i * DDIM, buf2, N);
        gin_bn_zero_kernel<<<1, 256, 0, stream>>>(stats);
        gin_bn_stats_kernel<<<statsGrid, 256, 0, stream>>>(buf2, N, rowsPerBlk, stats);
        gin_bn_apply_relu_kernel<<<ewGrid, 256, 0, stream>>>(
            buf2, stats, g2s + i * DDIM, be2s + i * DDIM, buf0, N);
    }

    // out = x[mask] @ W_out^T + b_out
    gin_out_kernel<<<(M * C + 255) / 256, 256, 0, stream>>>(
        buf0, mask, W_out, b_out, (float*)d_out, M, C);
}
